// MultiHeadAttention_17746804867465
// MI455X (gfx1250) — matrix operand order repaired and checked by reading, not compiled, not run on hardware
//
#include <hip/hip_runtime.h>
#include <hip/hip_bf16.h>

// ---------------------------------------------------------------------------
// MI455X (gfx1250) multi-head attention forward, bf16 WMMA pipeline.
//   B=2, S=2048, HIDDEN=1024, HEADS=16, HEAD_DIM=64
// ~69 GFLOP vs ~80MB unique traffic @ 23.3TB/s -> compute-bound -> all matmuls
// via v_wmma_f32_16x16x32_bf16 (f32 accumulate). This revision adds:
//  * 32x64 output tile per GEMM wave (2x arithmetic intensity vs 16x64)
//  * software-pipelined (double-buffered) fragment loads in GEMM + attention
//  * vectorized mask loads (b128) and vectorized f32->bf16 casts
// ---------------------------------------------------------------------------

typedef __attribute__((ext_vector_type(16))) __bf16 v16bf;
typedef __attribute__((ext_vector_type(8)))  float  v8f;
typedef __attribute__((ext_vector_type(4)))  float  v4f;
typedef __attribute__((ext_vector_type(4)))  unsigned int v4u;
typedef __attribute__((ext_vector_type(2)))  unsigned int v2u;

#define HIDDEN   1024
#define HEADS    16
#define HEAD_DIM 64
#define BATCH    2
#define SEQ      2048
#define MROWS    (BATCH * SEQ)          // 4096 token rows

static __device__ __forceinline__ unsigned short f32_to_bf16(float f) {
  unsigned int u = __float_as_uint(f);
  u += 0x7FFFu + ((u >> 16) & 1u);      // round-to-nearest-even
  return (unsigned short)(u >> 16);
}

union FragU { v4u u[2]; v16bf v; unsigned short s[16]; };

// A-operand (16xK=32, bf16): lane(l,h) holds row l, K = {8h+0..7, 16+8h+0..7}
static __device__ __forceinline__ v16bf load_fragA(const unsigned short* p, int ld,
                                                   int row0, int k0, int l, int h) {
  const unsigned short* r = p + (size_t)(row0 + l) * ld + k0 + h * 8;
  FragU t;
  t.u[0] = *(const v4u*)(r);
  t.u[1] = *(const v4u*)(r + 16);
  return t.v;
}

// B-operand (K=32 x 16, bf16): lane(l,h) holds col l, K = {8h+0..7, 16+8h+0..7}.
// Memory holds B as [col, K] row-major (i.e. W[n,k], Q[q,d], Vt[d,s]).
static __device__ __forceinline__ v16bf load_fragB(const unsigned short* p, int ld,
                                                   int col0, int k0, int l, int h) {
  const unsigned short* r = p + (size_t)(col0 + l) * ld + k0 + h * 8;
  FragU t;
  t.u[0] = *(const v4u*)(r);
  t.u[1] = *(const v4u*)(r + 16);
  return t.v;
}

static __device__ __forceinline__ v8f wmma_bf16(v16bf a, v16bf b, v8f c) {
  return __builtin_amdgcn_wmma_f32_16x16x32_bf16(false, a, false, b,
                                                 (short)0, c, false, false);
}

// ---------------------------------------------------------------------------
// Vectorized cast: 4 floats (b128 in) -> 4 bf16 (b64 out) per thread.
__global__ __launch_bounds__(256)
void cast_f32_bf16_x4(const float* __restrict__ src,
                      unsigned short* __restrict__ dst, int n4) {
  int i = blockIdx.x * blockDim.x + threadIdx.x;
  if (i < n4) {
    v4f f = *(const v4f*)(src + (size_t)i * 4);
    v2u o;
    o[0] = (unsigned int)f32_to_bf16(f[0]) | ((unsigned int)f32_to_bf16(f[1]) << 16);
    o[1] = (unsigned int)f32_to_bf16(f[2]) | ((unsigned int)f32_to_bf16(f[3]) << 16);
    *(v2u*)(dst + (size_t)i * 4) = o;
  }
}

// ---------------------------------------------------------------------------
// y = A(bf16)[M=4096,K=1024] @ W(bf16)[N=1024,K=1024]^T + bias.
// One 32x64 strip per wave: 2 A-frags x 4 B-frags -> 8 WMMAs per k-step,
// with next-k fragments prefetched (wrap-indexed, branch-free) before the
// current WMMAs so loads overlap the MMA pipe.
// mode 0: f32 row-major [M,N] -> outF
// mode 1: bf16 scatter to [B,H,S,64]  (Q / K)
// mode 2: bf16 scatter to [B,H,64,S]  (V transposed)
__global__ __launch_bounds__(256)
void gemm_bf16_wmma(const unsigned short* __restrict__ A,
                    const unsigned short* __restrict__ Bw,
                    const float* __restrict__ bias,
                    float* __restrict__ outF,
                    unsigned short* __restrict__ outH,
                    int mode) {
  const int lane = threadIdx.x & 31;
  const int wave = threadIdx.x >> 5;
  const int l = lane & 15, h = lane >> 4;
  const int ntiles = HIDDEN / 64;                  // 16 N-strips
  const int tid = blockIdx.x * 8 + wave;           // 2048 waves total
  const int m0 = (tid / ntiles) * 32;
  const int n0 = (tid % ntiles) * 64;

  v8f acc[8] = {};

  // Prologue: k=0 fragments.
  v16bf a0 = load_fragA(A, HIDDEN, m0,      0, l, h);
  v16bf a1 = load_fragA(A, HIDDEN, m0 + 16, 0, l, h);
  v16bf b0 = load_fragB(Bw, HIDDEN, n0 +  0, 0, l, h);
  v16bf b1 = load_fragB(Bw, HIDDEN, n0 + 16, 0, l, h);
  v16bf b2 = load_fragB(Bw, HIDDEN, n0 + 32, 0, l, h);
  v16bf b3 = load_fragB(Bw, HIDDEN, n0 + 48, 0, l, h);

  for (int k0 = 0; k0 < HIDDEN; k0 += 32) {
    const int kn = (k0 + 32) & (HIDDEN - 1);       // wrap: last prefetch benign
    const v16bf ca0 = a0, ca1 = a1;
    const v16bf cb0 = b0, cb1 = b1, cb2 = b2, cb3 = b3;
    // Prefetch next k-step while the 8 WMMAs below drain.
    a0 = load_fragA(A, HIDDEN, m0,      kn, l, h);
    a1 = load_fragA(A, HIDDEN, m0 + 16, kn, l, h);
    b0 = load_fragB(Bw, HIDDEN, n0 +  0, kn, l, h);
    b1 = load_fragB(Bw, HIDDEN, n0 + 16, kn, l, h);
    b2 = load_fragB(Bw, HIDDEN, n0 + 32, kn, l, h);
    b3 = load_fragB(Bw, HIDDEN, n0 + 48, kn, l, h);

    acc[0] = wmma_bf16(ca0, cb0, acc[0]);
    acc[1] = wmma_bf16(ca0, cb1, acc[1]);
    acc[2] = wmma_bf16(ca0, cb2, acc[2]);
    acc[3] = wmma_bf16(ca0, cb3, acc[3]);
    acc[4] = wmma_bf16(ca1, cb0, acc[4]);
    acc[5] = wmma_bf16(ca1, cb1, acc[5]);
    acc[6] = wmma_bf16(ca1, cb2, acc[6]);
    acc[7] = wmma_bf16(ca1, cb3, acc[7]);
  }

#pragma unroll
  for (int mh = 0; mh < 2; ++mh) {
#pragma unroll
    for (int s = 0; s < 4; ++s) {
#pragma unroll
      for (int e = 0; e < 8; ++e) {
        const int row = m0 + mh * 16 + e + 8 * h;  // C layout: M = e + 8h
        const int col = n0 + s * 16 + l;           //           N = lane&15
        const float v = acc[mh * 4 + s][e] + bias[col];
        if (mode == 0) {
          outF[(size_t)row * HIDDEN + col] = v;
        } else {
          const int b = row / SEQ, sq = row % SEQ;
          const int head = col >> 6, d = col & 63;
          size_t dst;
          if (mode == 1)
            dst = (((size_t)(b * HEADS + head)) * SEQ + sq) * HEAD_DIM + d;
          else
            dst = (((size_t)(b * HEADS + head)) * HEAD_DIM + d) * SEQ + sq;
          outH[dst] = f32_to_bf16(v);
        }
      }
    }
  }
}

// ---------------------------------------------------------------------------
// Flash attention, one 16-query tile per wave, 32-key chunks.
// Scores computed transposed (S^T = K-tile x Q^T) so the softmaxed tile maps
// element-wise into the P A-fragment of the PV WMMA (no cross-lane transpose).
// Next-chunk K frags + current-chunk V frags are issued before the softmax
// VALU work so their latency hides under exp/shuffle math.
__global__ __launch_bounds__(256)
void attn_flash_wmma(const unsigned short* __restrict__ Qm,
                     const unsigned short* __restrict__ Km,
                     const unsigned short* __restrict__ Vt,
                     const int* __restrict__ mask,
                     unsigned short* __restrict__ ctx) {
  const int lane = threadIdx.x & 31;
  const int wave = threadIdx.x >> 5;
  const int l = lane & 15, h = lane >> 4;
  const int bh = blockIdx.x;
  const int b = bh / HEADS, head = bh % HEADS;
  const int q0 = (blockIdx.y * 8 + wave) * 16;

  const unsigned short* Qp = Qm + (size_t)bh * SEQ * HEAD_DIM;
  const unsigned short* Kp = Km + (size_t)bh * SEQ * HEAD_DIM;
  const unsigned short* Vp = Vt + (size_t)bh * HEAD_DIM * SEQ;
  const int* mrow = mask + ((size_t)b * SEQ + (q0 + l)) * SEQ;  // this lane's q

  // Q as B-operand (B[d, q]): lane l = column q0+l, contiguous d. Resident.
  const v16bf qb0 = load_fragB(Qp, HEAD_DIM, q0, 0, l, h);
  const v16bf qb1 = load_fragB(Qp, HEAD_DIM, q0, 32, l, h);

  // Prologue: K fragments for chunk 0 (keys 0..31, A-operand, rows = keys).
  v16bf ka0 = load_fragA(Kp, HEAD_DIM, 0,  0,  l, h);
  v16bf ka1 = load_fragA(Kp, HEAD_DIM, 0,  32, l, h);
  v16bf ka2 = load_fragA(Kp, HEAD_DIM, 16, 0,  l, h);
  v16bf ka3 = load_fragA(Kp, HEAD_DIM, 16, 32, l, h);

  v8f acc[4] = {};
  float m_run = -3.0e38f, l_run = 0.0f;

  for (int j0 = 0; j0 < SEQ; j0 += 32) {
    // Mask values for this chunk: 4 aligned b128 loads (issued early).
    const v4u mk0a = *(const v4u*)(mrow + j0 + 8 * h);
    const v4u mk0b = *(const v4u*)(mrow + j0 + 8 * h + 4);
    const v4u mk1a = *(const v4u*)(mrow + j0 + 16 + 8 * h);
    const v4u mk1b = *(const v4u*)(mrow + j0 + 16 + 8 * h + 4);

    // Two 16-key S^T tiles, contract over d=64 (2 WMMAs each).
    v8f st0 = {}, st1 = {};
    st0 = wmma_bf16(ka0, qb0, st0);
    st0 = wmma_bf16(ka1, qb1, st0);
    st1 = wmma_bf16(ka2, qb0, st1);
    st1 = wmma_bf16(ka3, qb1, st1);

    // Issue next-chunk K loads (wrap-indexed) + this chunk's V loads now, so
    // they drain under the softmax VALU work below.
    const int jn = (j0 + 32) & (SEQ - 1);
    ka0 = load_fragA(Kp, HEAD_DIM, jn,      0,  l, h);
    ka1 = load_fragA(Kp, HEAD_DIM, jn,      32, l, h);
    ka2 = load_fragA(Kp, HEAD_DIM, jn + 16, 0,  l, h);
    ka3 = load_fragA(Kp, HEAD_DIM, jn + 16, 32, l, h);
    const v16bf vb0 = load_fragB(Vp, SEQ,  0, j0, l, h);
    const v16bf vb1 = load_fragB(Vp, SEQ, 16, j0, l, h);
    const v16bf vb2 = load_fragB(Vp, SEQ, 32, j0, l, h);
    const v16bf vb3 = load_fragB(Vp, SEQ, 48, j0, l, h);

    // Scale 1/sqrt(64), mask, chunk max (per q = lane; reduce halves: xor16).
    float cmax = -3.0e38f;
#pragma unroll
    for (int e = 0; e < 8; ++e) {
      float v0 = st0[e] * 0.125f;
      float v1 = st1[e] * 0.125f;
      const unsigned int mm0 = (e < 4) ? mk0a[e & 3] : mk0b[e & 3];
      const unsigned int mm1 = (e < 4) ? mk1a[e & 3] : mk1b[e & 3];
      if (mm0 == 0u) v0 = -1.0e9f;
      if (mm1 == 0u) v1 = -1.0e9f;
      st0[e] = v0; st1[e] = v1;
      cmax = fmaxf(cmax, fmaxf(v0, v1));
    }
    cmax = fmaxf(cmax, __shfl_xor(cmax, 16, 32));
    const float m_new = fmaxf(m_run, cmax);
    const float corr = __expf(m_run - m_new);

    // exp -> P A-fragment: S^T C-elem e of tile t is exactly A-elem t*8+e.
    FragU pa;
    float psum = 0.0f;
#pragma unroll
    for (int e = 0; e < 8; ++e) {
      const float p0 = __expf(st0[e] - m_new);
      const float p1 = __expf(st1[e] - m_new);
      psum += p0 + p1;
      pa.s[e]     = f32_to_bf16(p0);
      pa.s[e + 8] = f32_to_bf16(p1);
    }
    psum += __shfl_xor(psum, 16, 32);
    l_run = l_run * corr + psum;
    m_run = m_new;

    // Rescale out accumulator: acc row q = e + 8h, factor lives at lane q.
#pragma unroll
    for (int e = 0; e < 8; ++e) {
      const float ce = __shfl(corr, e + 8 * h, 32);
#pragma unroll
      for (int s = 0; s < 4; ++s) acc[s][e] *= ce;
    }

    // out += P @ V ; V column d is row d of Vt (contiguous keys).
    acc[0] = wmma_bf16(pa.v, vb0, acc[0]);
    acc[1] = wmma_bf16(pa.v, vb1, acc[1]);
    acc[2] = wmma_bf16(pa.v, vb2, acc[2]);
    acc[3] = wmma_bf16(pa.v, vb3, acc[3]);
  }

  // Normalize and scatter to [B, S, HIDDEN] bf16 context.
#pragma unroll
  for (int e = 0; e < 8; ++e) {
    const float linv = 1.0f / __shfl(l_run, e + 8 * h, 32);
    const int q = q0 + e + 8 * h;
#pragma unroll
    for (int s = 0; s < 4; ++s) {
      const int d = s * 16 + l;
      const float o = acc[s][e] * linv;
      ctx[((size_t)(b * SEQ + q)) * HIDDEN + head * HEAD_DIM + d] = f32_to_bf16(o);
    }
  }
}

// ---------------------------------------------------------------------------
extern "C" void kernel_launch(void* const* d_in, const int* in_sizes, int n_in,
                              void* d_out, int out_size, void* d_ws, size_t ws_size,
                              hipStream_t stream) {
  (void)in_sizes; (void)n_in; (void)out_size; (void)ws_size;
  const float* x    = (const float*)d_in[0];
  const int*   mask = (const int*)  d_in[1];
  const float* Wq   = (const float*)d_in[2];
  const float* bq   = (const float*)d_in[3];
  const float* Wk   = (const float*)d_in[4];
  const float* bk   = (const float*)d_in[5];
  const float* Wv   = (const float*)d_in[6];
  const float* bv   = (const float*)d_in[7];
  const float* Wo   = (const float*)d_in[8];
  const float* bo   = (const float*)d_in[9];
  float* out = (float*)d_out;

  // Workspace layout (bf16 = ushort), 40 MiB total:
  const size_t NX = (size_t)MROWS * HIDDEN;   // 4 Mi elems
  const size_t NW = (size_t)HIDDEN * HIDDEN;  // 1 Mi elems
  unsigned short* ws  = (unsigned short*)d_ws;
  unsigned short* xb  = ws;                   // x bf16 [4096,1024]
  unsigned short* wqb = xb  + NX;
  unsigned short* wkb = wqb + NW;
  unsigned short* wvb = wkb + NW;
  unsigned short* wob = wvb + NW;
  unsigned short* Qb  = wob + NW;             // [B,H,S,64]
  unsigned short* Kb  = Qb  + NX;             // [B,H,S,64]
  unsigned short* Vtb = Kb  + NX;             // [B,H,64,S]
  unsigned short* ctx = xb;                   // reuse x region after QKV GEMMs

  cast_f32_bf16_x4<<<(int)(NX / 4 / 256), 256, 0, stream>>>(x,  xb,  (int)(NX / 4));
  cast_f32_bf16_x4<<<(int)(NW / 4 / 256), 256, 0, stream>>>(Wq, wqb, (int)(NW / 4));
  cast_f32_bf16_x4<<<(int)(NW / 4 / 256), 256, 0, stream>>>(Wk, wkb, (int)(NW / 4));
  cast_f32_bf16_x4<<<(int)(NW / 4 / 256), 256, 0, stream>>>(Wv, wvb, (int)(NW / 4));
  cast_f32_bf16_x4<<<(int)(NW / 4 / 256), 256, 0, stream>>>(Wo, wob, (int)(NW / 4));

  const int gemmBlocks = (MROWS / 32) * (HIDDEN / 64) / 8;  // 256
  gemm_bf16_wmma<<<gemmBlocks, 256, 0, stream>>>(xb, wqb, bq, nullptr, Qb,  1);
  gemm_bf16_wmma<<<gemmBlocks, 256, 0, stream>>>(xb, wkb, bk, nullptr, Kb,  1);
  gemm_bf16_wmma<<<gemmBlocks, 256, 0, stream>>>(xb, wvb, bv, nullptr, Vtb, 2);

  // 32 (b,h) x 16 groups of 8 waves x 16 queries = 2048 queries
  attn_flash_wmma<<<dim3(BATCH * HEADS, SEQ / (16 * 8)), 256, 0, stream>>>(
      Qb, Kb, Vtb, mask, ctx);

  gemm_bf16_wmma<<<gemmBlocks, 256, 0, stream>>>(ctx, wob, bo, out, nullptr, 0);
}
